// AttentionDecoder_1082331758637
// MI455X (gfx1250) — matrix operand order repaired and checked by reading, not compiled, not run on hardware
//
#include <hip/hip_runtime.h>
#include <stdint.h>

// ---------------------------------------------------------------------------
// Types for CDNA5 WMMA (gfx1250, wave32)
// ---------------------------------------------------------------------------
typedef __attribute__((ext_vector_type(16))) __bf16 bf16x16;
typedef __attribute__((ext_vector_type(8)))  float  floatx8;

#define NEG_INF (-3.4028234663852886e38f)

// ---------------------------------------------------------------------------
// fp32 -> bf16 conversion (weights converted once per launch; activations
// converted per step). Deterministic, no cross-call state.
// ---------------------------------------------------------------------------
__global__ void f32_to_bf16_kernel(const float* __restrict__ in,
                                   __bf16* __restrict__ out, size_t n) {
  size_t i = (size_t)blockIdx.x * blockDim.x + threadIdx.x;
  if (i < n) out[i] = (__bf16)in[i];
}

// ---------------------------------------------------------------------------
// Full-M-blocked WMMA GEMM for M == 64:
//   C[64,N] (f32) = A[64,K] (bf16) @ W[N,K]^T (bf16) + bias
//
// One wave owns a full 64x16 column strip (all 4 M-tiles). Each B (weight)
// fragment is loaded from L2 exactly ONCE and fed to 4 WMMAs -> weight
// traffic is minimal (this is the binding resource at M=64).
//
// Fragment layouts per CDNA5 ISA 7.12.2:
//  A (16x32 bf16): lane half h: VGPR0-3 <- K = h*8 .. h*8+7,
//                               VGPR4-7 <- K = 16+h*8 .. 16+h*8+7   (M = lane%16)
//  B (32x16 bf16): lane half h: K = h*8..h*8+7, 16+h*8..16+h*8+7, same as A (N = lane%16)
//                  -> since B = W^T, lane loads two 8-element runs of W row N.
//  D (16x16 f32):  VGPR v -> row = v + 8*(lane/16), col = lane%16
// ---------------------------------------------------------------------------
__global__ void wmma_gemm_bf16_m64(const __bf16* __restrict__ A,
                                   const __bf16* __restrict__ W,
                                   const float* __restrict__ bias,
                                   float* __restrict__ C,
                                   int N, int K) {
  const int wave = blockIdx.x * (blockDim.x >> 5) + (threadIdx.x >> 5);
  const int tilesN = N >> 4;
  if (wave >= tilesN) return;
  const int tileN = wave;

  const int lane = threadIdx.x & 31;
  const int half = lane >> 4;   // 0: lanes 0-15, 1: lanes 16-31
  const int l16  = lane & 15;

  const __bf16* __restrict__ Wrow = W + (size_t)(tileN * 16 + l16) * K;
  const __bf16* __restrict__ Abase = A + (size_t)l16 * K;

  floatx8 acc0 = {}, acc1 = {}, acc2 = {}, acc3 = {};

  for (int k0 = 0; k0 < K; k0 += 32) {
    // B fragment: 2x8 bf16 of weight row, K = h*8.. and 16+h*8.. (loaded once, used 4x)
    bf16x16 bfrag;
    {
      union { bf16x16 v; uint4 q[2]; } u;
      u.q[0] = *(const uint4*)(Wrow + k0 + half * 8);
      u.q[1] = *(const uint4*)(Wrow + k0 + 16 + half * 8);
      bfrag = u.v;
    }
    // A fragments for the 4 M-tiles
    bf16x16 a0, a1, a2, a3;
    {
      const __bf16* r;
      union { bf16x16 v; uint4 q[2]; } u;
      r = Abase + (size_t)0 * 16 * K;
      u.q[0] = *(const uint4*)(r + k0 + half * 8);
      u.q[1] = *(const uint4*)(r + k0 + 16 + half * 8);
      a0 = u.v;
      r = Abase + (size_t)1 * 16 * K;
      u.q[0] = *(const uint4*)(r + k0 + half * 8);
      u.q[1] = *(const uint4*)(r + k0 + 16 + half * 8);
      a1 = u.v;
      r = Abase + (size_t)2 * 16 * K;
      u.q[0] = *(const uint4*)(r + k0 + half * 8);
      u.q[1] = *(const uint4*)(r + k0 + 16 + half * 8);
      a2 = u.v;
      r = Abase + (size_t)3 * 16 * K;
      u.q[0] = *(const uint4*)(r + k0 + half * 8);
      u.q[1] = *(const uint4*)(r + k0 + 16 + half * 8);
      a3 = u.v;
    }
    acc0 = __builtin_amdgcn_wmma_f32_16x16x32_bf16(false, a0, false, bfrag,
                                                   (short)0, acc0, false, false);
    acc1 = __builtin_amdgcn_wmma_f32_16x16x32_bf16(false, a1, false, bfrag,
                                                   (short)0, acc1, false, false);
    acc2 = __builtin_amdgcn_wmma_f32_16x16x32_bf16(false, a2, false, bfrag,
                                                   (short)0, acc2, false, false);
    acc3 = __builtin_amdgcn_wmma_f32_16x16x32_bf16(false, a3, false, bfrag,
                                                   (short)0, acc3, false, false);
  }

  const int col = tileN * 16 + l16;
  const float b = bias ? bias[col] : 0.0f;
#pragma unroll
  for (int v = 0; v < 8; ++v) {
    const int row = v + 8 * half;
    C[(size_t)(row +  0) * N + col] = acc0[v] + b;
    C[(size_t)(row + 16) * N + col] = acc1[v] + b;
    C[(size_t)(row + 32) * N + col] = acc2[v] + b;
    C[(size_t)(row + 48) * N + col] = acc3[v] + b;
  }
}

// ---------------------------------------------------------------------------
// x = concat(embed_table[ids_t], output)  -> bf16 [B, E+H]
// ---------------------------------------------------------------------------
__global__ void build_x_kernel(const int* __restrict__ ids_t,
                               const float* __restrict__ embed,
                               const float* __restrict__ output,
                               __bf16* __restrict__ x,
                               int B, int E, int H) {
  const int W = E + H;
  int idx = blockIdx.x * blockDim.x + threadIdx.x;
  if (idx >= B * W) return;
  int b = idx / W, j = idx % W;
  float v = (j < E) ? embed[(size_t)ids_t[b] * E + j]
                    : output[(size_t)b * H + (j - E)];
  x[idx] = (__bf16)v;
}

// ---------------------------------------------------------------------------
// GRU gate math (torch.nn.GRUCell):  writes hn (f32, in place ok) + hn bf16
// ---------------------------------------------------------------------------
__global__ void gru_pointwise_kernel(const float* __restrict__ gi,
                                     const float* __restrict__ gh,
                                     const float* __restrict__ h,
                                     float* __restrict__ hn,
                                     __bf16* __restrict__ hn_bf,
                                     int B, int H) {
  int idx = blockIdx.x * blockDim.x + threadIdx.x;
  if (idx >= B * H) return;
  int b = idx / H, j = idx % H;
  const float* gib = gi + (size_t)b * 3 * H;
  const float* ghb = gh + (size_t)b * 3 * H;
  float r = 1.0f / (1.0f + __expf(-(gib[j] + ghb[j])));
  float z = 1.0f / (1.0f + __expf(-(gib[H + j] + ghb[H + j])));
  float n = tanhf(gib[2 * H + j] + r * ghb[2 * H + j]);
  float o = (1.0f - z) * n + z * h[idx];
  hn[idx] = o;
  hn_bf[idx] = (__bf16)o;
}

// ---------------------------------------------------------------------------
// scores[b,s] = dot(q[b,:], context[b,s,:]) masked.  One wave per (b,s).
// ---------------------------------------------------------------------------
__global__ void attn_scores_kernel(const float* __restrict__ q,
                                   const float* __restrict__ context,
                                   const uint8_t* __restrict__ mask,
                                   float* __restrict__ scores,
                                   int B, int S, int H) {
  int wave = blockIdx.x * (blockDim.x >> 5) + (threadIdx.x >> 5);
  if (wave >= B * S) return;
  int b = wave / S;
  int lane = threadIdx.x & 31;
  const float* qb = q + (size_t)b * H;
  const float* cb = context + (size_t)wave * H;
  float s = 0.f;
  for (int j = lane; j < H; j += 32) s += qb[j] * cb[j];
  for (int off = 16; off; off >>= 1) s += __shfl_xor(s, off, 32);
  if (lane == 0) scores[wave] = mask[wave] ? s : NEG_INF;
}

// softmax over S=128 per row; blockDim == S == 128 (4 waves)
__global__ void softmax_rows_kernel(float* __restrict__ scores, int S) {
  __shared__ float red[4];
  int b = blockIdx.x, s = threadIdx.x, w = s >> 5;
  float v = scores[b * S + s];
  float m = v;
  for (int off = 16; off; off >>= 1) m = fmaxf(m, __shfl_xor(m, off, 32));
  if ((s & 31) == 0) red[w] = m;
  __syncthreads();
  m = fmaxf(fmaxf(red[0], red[1]), fmaxf(red[2], red[3]));
  __syncthreads();
  float e = __expf(v - m);
  float t = e;
  for (int off = 16; off; off >>= 1) t += __shfl_xor(t, off, 32);
  if ((s & 31) == 0) red[w] = t;
  __syncthreads();
  float tot = red[0] + red[1] + red[2] + red[3];
  scores[b * S + s] = e / tot;
}

// c[b,h] = sum_s attn[b,s] * context[b,s,h]
__global__ void attn_combine_kernel(const float* __restrict__ attn,
                                    const float* __restrict__ context,
                                    float* __restrict__ c,
                                    int B, int S, int H) {
  int idx = blockIdx.x * blockDim.x + threadIdx.x;
  if (idx >= B * H) return;
  int b = idx / H, h = idx % H;
  const float* ab = attn + (size_t)b * S;
  const float* cb = context + (size_t)b * S * H + h;
  float s = 0.f;
  for (int si = 0; si < S; ++si) s += ab[si] * cb[(size_t)si * H];
  c[idx] = s;
}

// cc = concat(c, h1) -> bf16 [B, 2H]
__global__ void build_cc_kernel(const float* __restrict__ c,
                                const float* __restrict__ h1,
                                __bf16* __restrict__ cc, int B, int H) {
  int W = 2 * H;
  int idx = blockIdx.x * blockDim.x + threadIdx.x;
  if (idx >= B * W) return;
  int b = idx / W, j = idx % W;
  float v = (j < H) ? c[(size_t)b * H + j] : h1[(size_t)b * H + (j - H)];
  cc[idx] = (__bf16)v;
}

// out = tanh(pre); dual write f32 + bf16
__global__ void tanh_store_kernel(const float* __restrict__ pre,
                                  float* __restrict__ outf,
                                  __bf16* __restrict__ outbf, int n) {
  int i = blockIdx.x * blockDim.x + threadIdx.x;
  if (i >= n) return;
  float v = tanhf(pre[i]);
  outf[i] = v;
  outbf[i] = (__bf16)v;
}

// log_softmax over V per row; blockDim = 256 (8 waves)
__global__ void log_softmax_rows_kernel(const float* __restrict__ logits,
                                        float* __restrict__ out, int V) {
  __shared__ float red[8];
  int b = blockIdx.x, w = threadIdx.x >> 5;
  const float* row = logits + (size_t)b * V;
  float* orow = out + (size_t)b * V;
  float m = NEG_INF;
  for (int j = threadIdx.x; j < V; j += blockDim.x) m = fmaxf(m, row[j]);
  for (int off = 16; off; off >>= 1) m = fmaxf(m, __shfl_xor(m, off, 32));
  if ((threadIdx.x & 31) == 0) red[w] = m;
  __syncthreads();
  float mx = red[0];
#pragma unroll
  for (int i = 1; i < 8; ++i) mx = fmaxf(mx, red[i]);
  __syncthreads();
  float s = 0.f;
  for (int j = threadIdx.x; j < V; j += blockDim.x) s += __expf(row[j] - mx);
  for (int off = 16; off; off >>= 1) s += __shfl_xor(s, off, 32);
  if ((threadIdx.x & 31) == 0) red[w] = s;
  __syncthreads();
  float tot = 0.f;
#pragma unroll
  for (int i = 0; i < 8; ++i) tot += red[i];
  float lse = mx + __logf(tot);
  for (int j = threadIdx.x; j < V; j += blockDim.x) orow[j] = row[j] - lse;
}

// ---------------------------------------------------------------------------
// Host-side launcher
// ---------------------------------------------------------------------------
extern "C" void kernel_launch(void* const* d_in, const int* in_sizes, int n_in,
                              void* d_out, int out_size, void* d_ws, size_t ws_size,
                              hipStream_t stream) {
  (void)in_sizes; (void)n_in; (void)out_size; (void)ws_size;
  const int T = 32, B = 64, S = 128, H = 1024, E = 512, V = 32000;
  const int H3 = 3 * H;

  const int*   ids       = (const int*)  d_in[0];
  const float* embed     = (const float*)d_in[1];
  const float* W_ih0     = (const float*)d_in[2];
  const float* W_hh0     = (const float*)d_in[3];
  const float* b_ih0     = (const float*)d_in[4];
  const float* b_hh0     = (const float*)d_in[5];
  const float* W_ih1     = (const float*)d_in[6];
  const float* W_hh1     = (const float*)d_in[7];
  const float* b_ih1     = (const float*)d_in[8];
  const float* b_hh1     = (const float*)d_in[9];
  const float* attn_Win  = (const float*)d_in[10];
  const float* attn_Wout = (const float*)d_in[11];
  const float* gen_W     = (const float*)d_in[12];
  const float* gen_b     = (const float*)d_in[13];
  const float* hidden    = (const float*)d_in[14];
  const float* context   = (const float*)d_in[15];
  const uint8_t* cmask   = (const uint8_t*)d_in[16];
  const float* prev_out  = (const float*)d_in[17];

  float* out = (float*)d_out;

  // ---- workspace layout ----
  char* ws = (char*)d_ws;
  size_t off = 0;
  auto alloc = [&](size_t bytes) -> char* {
    char* p = ws + off;
    off = (off + bytes + 255) & ~(size_t)255;
    return p;
  };
  __bf16* W_ih0_bf  = (__bf16*)alloc((size_t)H3 * (E + H) * 2);
  __bf16* W_hh0_bf  = (__bf16*)alloc((size_t)H3 * H * 2);
  __bf16* W_ih1_bf  = (__bf16*)alloc((size_t)H3 * H * 2);
  __bf16* W_hh1_bf  = (__bf16*)alloc((size_t)H3 * H * 2);
  __bf16* Win_bf    = (__bf16*)alloc((size_t)H * H * 2);
  __bf16* Wout_bf   = (__bf16*)alloc((size_t)H * 2 * H * 2);
  __bf16* genW_bf   = (__bf16*)alloc((size_t)V * H * 2);
  __bf16* x_bf      = (__bf16*)alloc((size_t)B * (E + H) * 2);
  __bf16* cc_bf     = (__bf16*)alloc((size_t)B * 2 * H * 2);
  __bf16* h0_bf     = (__bf16*)alloc((size_t)B * H * 2);
  __bf16* h1_bf     = (__bf16*)alloc((size_t)B * H * 2);
  __bf16* out_bf    = (__bf16*)alloc((size_t)B * H * 2);
  float*  gi        = (float*)alloc((size_t)B * H3 * 4);
  float*  gh        = (float*)alloc((size_t)B * H3 * 4);
  float*  h0_f      = (float*)alloc((size_t)B * H * 4);
  float*  h1_f      = (float*)alloc((size_t)B * H * 4);
  float*  out_f     = (float*)alloc((size_t)B * H * 4);
  float*  q_f       = (float*)alloc((size_t)B * H * 4);
  float*  scores    = (float*)alloc((size_t)B * S * 4);
  float*  c_f       = (float*)alloc((size_t)B * H * 4);
  float*  pre_f     = (float*)alloc((size_t)B * H * 4);
  float*  logits    = (float*)alloc((size_t)B * V * 4);

  auto cvt = [&](const float* src, __bf16* dst, size_t n) {
    int blocks = (int)((n + 511) / 512);
    f32_to_bf16_kernel<<<blocks, 512, 0, stream>>>(src, dst, n);
  };
  // M is always 64 here: full-M-blocked GEMM, one wave per 16-wide N strip.
  auto gemm = [&](const __bf16* A, const __bf16* W, const float* bias,
                  float* C, int N, int K) {
    int waves = N / 16;
    int blocks = (waves + 7) / 8;
    wmma_gemm_bf16_m64<<<blocks, 256, 0, stream>>>(A, W, bias, C, N, K);
  };

  // ---- one-time-per-launch: weights -> bf16 (L2-resident thereafter) ----
  cvt(W_ih0,     W_ih0_bf, (size_t)H3 * (E + H));
  cvt(W_hh0,     W_hh0_bf, (size_t)H3 * H);
  cvt(W_ih1,     W_ih1_bf, (size_t)H3 * H);
  cvt(W_hh1,     W_hh1_bf, (size_t)H3 * H);
  cvt(attn_Win,  Win_bf,   (size_t)H * H);
  cvt(attn_Wout, Wout_bf,  (size_t)H * 2 * H);
  cvt(gen_W,     genW_bf,  (size_t)V * H);

  // ---- recurrent state init ----
  hipMemcpyAsync(h0_f, hidden,             (size_t)B * H * 4, hipMemcpyDeviceToDevice, stream);
  hipMemcpyAsync(h1_f, hidden + (size_t)B * H, (size_t)B * H * 4, hipMemcpyDeviceToDevice, stream);
  hipMemcpyAsync(out_f, prev_out,          (size_t)B * H * 4, hipMemcpyDeviceToDevice, stream);
  cvt(h0_f, h0_bf, (size_t)B * H);
  cvt(h1_f, h1_bf, (size_t)B * H);

  const int BH = B * H;
  for (int t = 0; t < T; ++t) {
    // input feeding: x = [emb_t, output]
    build_x_kernel<<<(B * (E + H) + 255) / 256, 256, 0, stream>>>(
        ids + (size_t)t * B, embed, out_f, x_bf, B, E, H);

    // GRU layer 0
    gemm(x_bf,  W_ih0_bf, b_ih0, gi, H3, E + H);
    gemm(h0_bf, W_hh0_bf, b_hh0, gh, H3, H);
    gru_pointwise_kernel<<<(BH + 255) / 256, 256, 0, stream>>>(
        gi, gh, h0_f, h0_f, h0_bf, B, H);

    // GRU layer 1
    gemm(h0_bf, W_ih1_bf, b_ih1, gi, H3, H);
    gemm(h1_bf, W_hh1_bf, b_hh1, gh, H3, H);
    gru_pointwise_kernel<<<(BH + 255) / 256, 256, 0, stream>>>(
        gi, gh, h1_f, h1_f, h1_bf, B, H);

    // global attention ('general')
    gemm(h1_bf, Win_bf, nullptr, q_f, H, H);
    attn_scores_kernel<<<(B * S + 7) / 8, 256, 0, stream>>>(
        q_f, context, cmask, scores, B, S, H);
    softmax_rows_kernel<<<B, S, 0, stream>>>(scores, S);
    attn_combine_kernel<<<(BH + 255) / 256, 256, 0, stream>>>(
        scores, context, c_f, B, S, H);
    build_cc_kernel<<<(B * 2 * H + 255) / 256, 256, 0, stream>>>(
        c_f, h1_f, cc_bf, B, H);
    gemm(cc_bf, Wout_bf, nullptr, pre_f, H, 2 * H);
    tanh_store_kernel<<<(BH + 255) / 256, 256, 0, stream>>>(
        pre_f, out_f, out_bf, BH);

    // generator + log_softmax -> d_out[t]
    gemm(out_bf, genW_bf, gen_b, logits, V, H);
    log_softmax_rows_kernel<<<B, 256, 0, stream>>>(
        logits, out + (size_t)t * B * V, V);
  }

  // tail outputs: stacked hidden [2,B,H] then output [B,H]
  size_t tail = (size_t)T * B * V;
  hipMemcpyAsync(out + tail,              h0_f,  (size_t)BH * 4, hipMemcpyDeviceToDevice, stream);
  hipMemcpyAsync(out + tail + BH,         h1_f,  (size_t)BH * 4, hipMemcpyDeviceToDevice, stream);
  hipMemcpyAsync(out + tail + 2 * (size_t)BH, out_f, (size_t)BH * 4, hipMemcpyDeviceToDevice, stream);
}
